// BatchHardTripletLoss_27118423507255
// MI455X (gfx1250) — matrix operand order repaired and checked by reading, not compiled, not run on hardware
//
#include <hip/hip_runtime.h>
#include <hip/hip_bf16.h>
#include <math.h>

// ---------------------------------------------------------------------------
// Batch-hard triplet loss, MI455X (gfx1250, wave32, WMMA + TDM).
//   B = 4096 anchors, D = 2048 dims, 128 labels.
//   Gram matrix via v_wmma_f32_16x16x32_bf16 (f32 accumulate), mining fused.
//   B-block staged into LDS by the Tensor Data Mover (double-buffered).
// ---------------------------------------------------------------------------

typedef __attribute__((ext_vector_type(16))) __bf16 v16bf;
typedef __attribute__((ext_vector_type(8)))  __bf16 v8bf;
typedef __attribute__((ext_vector_type(8)))  float  v8f;
typedef __attribute__((ext_vector_type(4)))  unsigned int u32x4;
typedef __attribute__((ext_vector_type(8)))  int    i32x8;
typedef __attribute__((ext_vector_type(4)))  int    i32x4;

#define BATCH   4096
#define DIM     2048
#define MB      128          // rows per workgroup (mine kernel)
#define NB      128          // cols per N-tile
#define KC      32           // K per WMMA chunk
#define NKC     (DIM / KC)   // 64 K-chunks
#define MARGIN  0.3f
#define EPS     1e-6f
#define BIGV    1000000.0f

#if defined(__HIP_DEVICE_COMPILE__) && __has_builtin(__builtin_amdgcn_tensor_load_to_lds)
#define USE_TDM 1
#else
#define USE_TDM 0
#endif

// ---------------------------------------------------------------------------
// Kernel 1: f32 -> bf16 quantize + row |x|^2 (computed on quantized values so
// d2 = sq[i]+sq[j]-2*dot stays non-negative-consistent with the bf16 dot).
// ---------------------------------------------------------------------------
__global__ __launch_bounds__(256)
void k_prep(const float* __restrict__ emb, __bf16* __restrict__ embB,
            float* __restrict__ sq) {
    const int row = blockIdx.x;
    const int tid = threadIdx.x;
    float s = 0.f;
    for (int k = tid; k < DIM; k += 256) {
        float v  = emb[(size_t)row * DIM + k];
        __bf16 b = (__bf16)v;
        embB[(size_t)row * DIM + k] = b;
        float fb = (float)b;
        s += fb * fb;
    }
    __shared__ float red[256];
    red[tid] = s;
    __syncthreads();
    #pragma unroll
    for (int off = 128; off > 0; off >>= 1) {
        if (tid < off) red[tid] += red[tid + off];
        __syncthreads();
    }
    if (tid == 0) sq[row] = red[0];
}

#if USE_TDM
// ---------------------------------------------------------------------------
// TDM: DMA a 128-row x 32-k bf16 tile (row stride DIM) into LDS at lds_off,
// packed row-contiguously (64 B per row) == our [col][k] fragment layout.
// Descriptor packing per CDNA5 ISA:
//   g0: count=1 | lds_addr | global_addr[56:0] | type=2
//   g1: data_size=2B | tensor_dim0=DIM | tensor_dim1=BATCH | tile 32x128
//       | tensor_dim0_stride=DIM ; groups 2/3 zero (2D tile)
// ---------------------------------------------------------------------------
__device__ __forceinline__ void tdm_load_B(const __bf16* gptr, unsigned lds_off) {
    unsigned long long ga = (unsigned long long)(uintptr_t)gptr;
    u32x4 g0;
    g0[0] = 1u;                                   // count=1 (valid descriptor)
    g0[1] = lds_off;                              // lds_addr (bytes)
    g0[2] = (unsigned)ga;                         // global_addr[31:0]
    g0[3] = (unsigned)((ga >> 32) & 0x1FFFFFFu)   // global_addr[56:32]
            | 0x80000000u;                        // type=2 ("image")
    i32x8 g1;
    g1[0] = 0x00010000;                           // data_size=1 -> 2 bytes
    g1[1] = (int)((DIM & 0xFFFF) << 16);          // tensor_dim0 low16 @ [63:48]
    g1[2] = (int)(((DIM >> 16) & 0xFFFF)          // tensor_dim0 hi @ [79:64]
            | ((BATCH & 0xFFFF) << 16));          // tensor_dim1 low16 @ [111:80]
    g1[3] = (int)(((BATCH >> 16) & 0xFFFF)        // tensor_dim1 hi
            | (KC << 16));                        // tile_dim0=32 @ [127:112]
    g1[4] = NB;                                   // tile_dim1=128 @ [143:128]
    g1[5] = DIM;                                  // tensor_dim0_stride low32
    g1[6] = 0;
    g1[7] = 0;
    i32x4 gz = (i32x4)0;
#if defined(__clang_major__) && (__clang_major__ >= 23)
    __builtin_amdgcn_tensor_load_to_lds(g0, g1, gz, gz, (i32x8)0, 0);
#else
    __builtin_amdgcn_tensor_load_to_lds(g0, g1, gz, gz, 0);
#endif
}
#endif

// ---------------------------------------------------------------------------
// Kernel 2: fused Gram-matrix + batch-hard mining.
// Workgroup = 8 waves, owns 128 rows. Wave w computes rows [w*16, w*16+16).
// For each 128-col N-tile: accumulate full K=2048 via WMMA (B-block DMA'd to
// LDS by TDM, double buffered; A fragment loaded from global in contiguous-k
// layout), then convert dots to distances and update hard-pos / hard-neg.
// ---------------------------------------------------------------------------
__global__ __launch_bounds__(256)
void k_mine(const __bf16* __restrict__ embB, const float* __restrict__ sq,
            const int* __restrict__ labels, int* __restrict__ hp,
            int* __restrict__ hn) {
    const int tid    = threadIdx.x;
    const int wave   = tid >> 5;
    const int lane   = tid & 31;
    const int laneLo = lane & 15;
    const int laneHi = lane >> 4;

    const int m0      = blockIdx.x * MB;
    // C/D layout: VGPR r holds row r (lanes 0-15) / row r+8 (lanes 16-31)
    const int rowBase = m0 + wave * 16 + laneHi * 8;
    // A layout: lanes 0-15 and 16-31 both hold rows M=0..15; laneHi picks K half
    const int aRow    = m0 + wave * 16 + laneLo;

    float rowSq[8];
    int   rowLab[8];
    #pragma unroll
    for (int r = 0; r < 8; ++r) {
        rowSq[r]  = sq[rowBase + r];
        rowLab[r] = labels[rowBase + r];
    }

    float bestPos[8], bestNeg[8];
    int   bpi[8], bni[8];
    #pragma unroll
    for (int r = 0; r < 8; ++r) {
        bestPos[r] = -1.0f; bpi[r] = -1;      // non-positives compare as -1
        bestNeg[r] = 3.0e38f; bni[r] = 0;
    }

    // Double-buffered B-block: 2 x (128 cols x 32 k), [col][k] contiguous-k
    __shared__ alignas(32) __bf16 Bsh[2][NB * KC];

#if USE_TDM
    const unsigned ldsOff0 = (unsigned)(uintptr_t)(void*)&Bsh[0][0];
    const unsigned ldsOff1 = (unsigned)(uintptr_t)(void*)&Bsh[1][0];
#else
    const int fCol  = tid >> 1;   // 0..127
    const int fHalf = tid & 1;    // which 16-element k half this thread stages
#endif

    for (int n0 = 0; n0 < BATCH; n0 += NB) {
        v8f acc[8];
        #pragma unroll
        for (int t = 0; t < 8; ++t) acc[t] = v8f{};

#if USE_TDM
        if (tid < 32)  // wave 0 issues the DMA for chunk 0 of this N-tile
            tdm_load_B(embB + (size_t)n0 * DIM, ldsOff0);
#endif
        for (int i = 0; i < NKC; ++i) {
            const int kc  = i * KC;
            const int cur = i & 1;
#if USE_TDM
            if (tid < 32) __builtin_amdgcn_s_wait_tensorcnt(0); // buf[cur] ready
            __syncthreads();   // publish buf[cur]; all done reading buf[1-cur]
            if (tid < 32 && (i + 1) < NKC)  // DMA next chunk under the WMMAs
                tdm_load_B(embB + (size_t)n0 * DIM + kc + KC,
                           cur ? ldsOff0 : ldsOff1);
#else
            __syncthreads();   // previous chunk's LDS reads done
            {
                const v8bf* g = (const v8bf*)(embB + (size_t)(n0 + fCol) * DIM
                                              + kc + fHalf * 16);
                v8bf t0 = g[0];
                v8bf t1 = g[1];
                v8bf* s = (v8bf*)(&Bsh[cur][fCol * KC + fHalf * 16]);
                s[0] = t0;
                s[1] = t1;
            }
            __syncthreads();
#endif
            // A fragment (16x32 bf16): laneHi==0 -> K {0..15},
            //                          laneHi==1 -> K {16..31}
            const __bf16* abase = embB + (size_t)aRow * DIM + kc + laneHi * 16;
            v8bf alo = *(const v8bf*)(abase);
            v8bf ahi = *(const v8bf*)(abase + 8);
            v16bf afrag;
            #pragma unroll
            for (int q = 0; q < 8; ++q) {
                afrag[q]     = alo[q];
                afrag[q + 8] = ahi[q];
            }

            #pragma unroll
            for (int t = 0; t < 8; ++t) {
                // B fragment (32x16 bf16): lane holds col t*16+laneLo,
                // 16 contiguous K starting at laneHi*16
                const v16bf bfrag = *(const v16bf*)(
                    &Bsh[cur][(t * 16 + laneLo) * KC + laneHi * 16]);
                acc[t] = __builtin_amdgcn_wmma_f32_16x16x32_bf16(
                    false, afrag, false, bfrag, (short)0, acc[t], false, false);
            }
        }

        // distances + running hard-pos / hard-neg (no LDS touched here)
        #pragma unroll
        for (int t = 0; t < 8; ++t) {
            const int   col  = n0 + t * 16 + laneLo;
            const float cSq  = sq[col];
            const int   cLab = labels[col];
            #pragma unroll
            for (int r = 0; r < 8; ++r) {
                float dot  = acc[t][r];
                float d2   = rowSq[r] + cSq - 2.0f * dot;
                float dist = sqrtf(fmaxf(d2, 0.0f));
                bool  same = (rowLab[r] == cLab);
                int   ridx = rowBase + r;
                float pc = (same && col != ridx) ? dist : -1.0f;
                if (pc > bestPos[r]) { bestPos[r] = pc; bpi[r] = col; }
                float nc = same ? dist + BIGV : dist;
                if (nc < bestNeg[r]) { bestNeg[r] = nc; bni[r] = col; }
            }
        }
    }

    // reduce across the 16 lanes of each half-wave (masks 1..8 stay in-half),
    // tie-break toward the smaller column index (mimics argmax/argmin).
    #pragma unroll
    for (int r = 0; r < 8; ++r) {
        #pragma unroll
        for (int m = 8; m >= 1; m >>= 1) {
            float oP  = __shfl_xor(bestPos[r], m);
            int   oPi = __shfl_xor(bpi[r], m);
            if (oP > bestPos[r] ||
                (oP == bestPos[r] && oPi >= 0 && (bpi[r] < 0 || oPi < bpi[r]))) {
                bestPos[r] = oP; bpi[r] = oPi;
            }
            float oN  = __shfl_xor(bestNeg[r], m);
            int   oNi = __shfl_xor(bni[r], m);
            if (oN < bestNeg[r] || (oN == bestNeg[r] && oNi < bni[r])) {
                bestNeg[r] = oN; bni[r] = oNi;
            }
        }
        if (laneLo == 0) {
            int row = rowBase + r;
            hp[row] = bpi[r];                                    // -1: no positive
            hn[row] = (bestNeg[r] >= 0.5f * BIGV) ? -1 : bni[r]; // -1: no negative
        }
    }
}

// ---------------------------------------------------------------------------
// Kernel 3: exact f32 recompute of dp/dn (matches torch TripletMarginLoss:
// eps added per-component before the norm) + hinge; one block per anchor.
// ---------------------------------------------------------------------------
__global__ __launch_bounds__(256)
void k_pairs(const float* __restrict__ emb, const int* __restrict__ hp,
             const int* __restrict__ hn, float* __restrict__ per,
             float* __restrict__ vflag) {
    const int i   = blockIdx.x;
    const int tid = threadIdx.x;
    const int p = hp[i], n = hn[i];
    const bool valid = (p >= 0) && (n >= 0);
    const int pp = valid ? p : 0;
    const int nn = valid ? n : 0;

    float dp = 0.f, dn = 0.f;
    for (int k = tid; k < DIM; k += 256) {
        float a = emb[(size_t)i * DIM + k];
        float x = a - emb[(size_t)pp * DIM + k] + EPS;
        float y = a - emb[(size_t)nn * DIM + k] + EPS;
        dp += x * x;
        dn += y * y;
    }
    __shared__ float rp[256], rn[256];
    rp[tid] = dp; rn[tid] = dn;
    __syncthreads();
    #pragma unroll
    for (int off = 128; off > 0; off >>= 1) {
        if (tid < off) { rp[tid] += rp[tid + off]; rn[tid] += rn[tid + off]; }
        __syncthreads();
    }
    if (tid == 0) {
        float l = sqrtf(rp[0]) - sqrtf(rn[0]) + MARGIN;
        per[i]   = valid ? fmaxf(l, 0.f) : 0.f;
        vflag[i] = valid ? 1.f : 0.f;
    }
}

// ---------------------------------------------------------------------------
// Kernel 4: deterministic single-block mean over valid anchors.
// ---------------------------------------------------------------------------
__global__ __launch_bounds__(256)
void k_reduce(const float* __restrict__ per, const float* __restrict__ vflag,
              float* __restrict__ out) {
    const int tid = threadIdx.x;
    float s = 0.f, c = 0.f;
    for (int j = tid; j < BATCH; j += 256) { s += per[j]; c += vflag[j]; }
    __shared__ float rs[256], rc[256];
    rs[tid] = s; rc[tid] = c;
    __syncthreads();
    #pragma unroll
    for (int off = 128; off > 0; off >>= 1) {
        if (tid < off) { rs[tid] += rs[tid + off]; rc[tid] += rc[tid + off]; }
        __syncthreads();
    }
    if (tid == 0) out[0] = (rc[0] > 0.f) ? rs[0] / rc[0] : 0.f;
}

// ---------------------------------------------------------------------------
extern "C" void kernel_launch(void* const* d_in, const int* in_sizes, int n_in,
                              void* d_out, int out_size, void* d_ws, size_t ws_size,
                              hipStream_t stream) {
    (void)in_sizes; (void)n_in; (void)out_size; (void)ws_size;
    const float* emb    = (const float*)d_in[0];
    const int*   labels = (const int*)d_in[1];
    float*       out    = (float*)d_out;

    // workspace layout (~16.9 MB)
    char* ws = (char*)d_ws;
    __bf16* embB  = (__bf16*)ws;                                   // 16 MiB
    size_t  off   = (size_t)BATCH * DIM * sizeof(__bf16);
    float*  sq    = (float*)(ws + off); off += BATCH * sizeof(float);
    int*    hp    = (int*)  (ws + off); off += BATCH * sizeof(int);
    int*    hn    = (int*)  (ws + off); off += BATCH * sizeof(int);
    float*  per   = (float*)(ws + off); off += BATCH * sizeof(float);
    float*  vflag = (float*)(ws + off);

    k_prep  <<<BATCH,      256, 0, stream>>>(emb, embB, sq);
    k_mine  <<<BATCH / MB, 256, 0, stream>>>(embB, sq, labels, hp, hn);
    k_pairs <<<BATCH,      256, 0, stream>>>(emb, hp, hn, per, vflag);
    k_reduce<<<1,          256, 0, stream>>>(per, vflag, out);
}
